// LLModel_858993459620
// MI455X (gfx1250) — matrix operand order repaired and checked by reading, not compiled, not run on hardware
//
#include <hip/hip_runtime.h>
#include <hip/hip_bf16.h>
#include <math.h>

// Model dims
#define V_  32000
#define D_  1024
#define H_  16
#define L_  4
#define F_  4096
#define S_  1024
#define B_  2
#define DK  64
#define MTOT (B_ * S_)   // 2048 rows of activations

// GEMM tiling
#define BM 128
#define BN 128
#define BK 32

typedef __bf16 bf16_t;
typedef __bf16        v16bf __attribute__((ext_vector_type(16)));
typedef float         v8f   __attribute__((ext_vector_type(8)));
typedef unsigned int  u32x4 __attribute__((ext_vector_type(4)));
typedef int           i32x4 __attribute__((ext_vector_type(4)));

typedef __attribute__((address_space(1))) i32x4 gv4i;
typedef __attribute__((address_space(3))) i32x4 lv4i;

// ---------------------------------------------------------------------------
// CDNA5 async copy: global -> LDS, 16 bytes per lane, tracked by ASYNCcnt.
// Builtin signature (probe-confirmed):
//   (int4 addrspace(1)* gsrc, int4 addrspace(3)* ldst, imm offset, imm cpol)
// ---------------------------------------------------------------------------
__device__ __forceinline__ void async_cp16(bf16_t* ldst, const bf16_t* gsrc) {
#if defined(__has_builtin) && __has_builtin(__builtin_amdgcn_global_load_async_to_lds_b128)
  __builtin_amdgcn_global_load_async_to_lds_b128((gv4i*)gsrc, (lv4i*)ldst, 0, 0);
#else
  const unsigned loff =
      (unsigned)(size_t)(__attribute__((address_space(3))) bf16_t*)ldst;
  asm volatile("global_load_async_to_lds_b128 %0, %1, off"
               :: "v"(loff), "v"(gsrc) : "memory");
#endif
}

__device__ __forceinline__ void wait_async0() {
#if defined(__has_builtin) && __has_builtin(__builtin_amdgcn_s_wait_asynccnt)
  __builtin_amdgcn_s_wait_asynccnt(0);
#else
  asm volatile("s_wait_asynccnt 0x0" ::: "memory");
#endif
}

// ---------------------------------------------------------------------------
// WMMA fragment loaders (CDNA5 16-bit layouts, CDNA5 ISA §7.12.2)
// A 16x32 (MxK): lane L<16 = row L, halves 0..7 -> K0..7, halves 8..15 -> K16..23
//                lane L+16 = row L, halves 0..7 -> K8..15, halves 8..15 -> K24..31
// B 32x16 (KxN): lane n<16 = col n, K0..7,16..23 ; lane n+16 = col n, K8..15,24..31
// C/D 16x16 f32: vgpr r, lanes0-15 -> M=r, lanes16-31 -> M=r+8 ; N = lane&15
// ---------------------------------------------------------------------------
__device__ __forceinline__ v16bf load_a_frag(const bf16_t* base, int ld,
                                             int row0, int k0) {
  const int lane = threadIdx.x & 31;
  const int m = lane & 15, hi = lane >> 4;
  const bf16_t* p = base + (size_t)(row0 + m) * ld + k0 + hi * 8;
  union { v16bf v; u32x4 q[2]; } u;
  u.q[0] = *(const u32x4*)(p);        // K = hi*8 + 0..7
  u.q[1] = *(const u32x4*)(p + 16);   // K = 16 + hi*8 + 0..7
  return u.v;
}

__device__ __forceinline__ v16bf load_b_frag(const bf16_t* base, int ld,
                                             int col0, int k0) {
  const int lane = threadIdx.x & 31;
  const int n = lane & 15, hi = lane >> 4;
  const bf16_t* p = base + (size_t)(col0 + n) * ld + k0 + hi * 8;
  union { v16bf v; u32x4 q[2]; } u;
  u.q[0] = *(const u32x4*)(p);        // K = hi*8 + 0..7
  u.q[1] = *(const u32x4*)(p + 16);   // K = 16 + hi*8 + 0..7
  return u.v;
}

__device__ __forceinline__ v8f wmma_bf16(v16bf a, v16bf b, v8f c) {
  return __builtin_amdgcn_wmma_f32_16x16x32_bf16(false, a, false, b, (short)0, c,
                                                 false, false);
}

// ---------------------------------------------------------------------------
// GEMM: C[M,N] = A[M,K] @ W[N,K]^T  (+ optional residual), A/W bf16, C f32.
// 256 threads = 8 waves. Block tile BM x BN, K-chunk BK, double-buffered LDS
// fed by async global->LDS copies; wave w owns rows [BM*bx + 16w, +16).
// All fragments are hoisted into registers before the WMMA chain so the
// ds_loads issue as one clause and the 8 WMMAs run back-to-back.
// ---------------------------------------------------------------------------
__global__ __launch_bounds__(256) void gemm_bf16_kernel(
    const bf16_t* __restrict__ A, const bf16_t* __restrict__ W,
    float* __restrict__ C, const float* __restrict__ Res,
    int M, int N, int K)
{
  __shared__ bf16_t As[2][BM * BK];   // 8 KB per buffer
  __shared__ bf16_t Ws[2][BN * BK];   // 8 KB per buffer

  const int tid = threadIdx.x;
  const int wave = tid >> 5;
  const int mblk = blockIdx.x * BM;
  const int nblk = blockIdx.y * BN;

  // Stage one BK-chunk of A (BM x BK) and W (BN x BK) into LDS buffer `buf`.
  // 512 x 16B chunks per tile; each thread issues 2 chunks per tile.
  auto stage = [&](int buf, int k0) {
    #pragma unroll
    for (int j = 0; j < 2; ++j) {
      const int c = tid + j * 256;
      const int row = c >> 2;           // 4 x 16B chunks per 32-half row
      const int col = (c & 3) * 8;
      async_cp16(&As[buf][row * BK + col],
                 A + (size_t)(mblk + row) * K + k0 + col);
      async_cp16(&Ws[buf][row * BK + col],
                 W + (size_t)(nblk + row) * K + k0 + col);
    }
  };

  v8f acc[8] = {v8f{}, v8f{}, v8f{}, v8f{}, v8f{}, v8f{}, v8f{}, v8f{}};

  stage(0, 0);
  wait_async0();
  __syncthreads();

  const int nk = K / BK;
  for (int kc = 0; kc < nk; ++kc) {
    const int cur = kc & 1;
    if (kc + 1 < nk) stage(cur ^ 1, (kc + 1) * BK);  // overlap DMA w/ compute

    const bf16_t* abase = &As[cur][0];
    const bf16_t* wbase = &Ws[cur][0];

    // hoist all fragment loads: one ds clause, one wait, then 8 WMMAs
    const v16bf a = load_a_frag(abase, BK, wave * 16, 0);
    v16bf b[8];
    #pragma unroll
    for (int t = 0; t < 8; ++t) b[t] = load_b_frag(wbase, BK, t * 16, 0);
    #pragma unroll
    for (int t = 0; t < 8; ++t) acc[t] = wmma_bf16(a, b[t], acc[t]);

    wait_async0();     // next buffer landed
    __syncthreads();   // everyone done reading `cur` before it is re-staged
  }

  const int lane = tid & 31;
  const int cn = lane & 15, hi = lane >> 4;
  #pragma unroll
  for (int t = 0; t < 8; ++t) {
    #pragma unroll
    for (int r = 0; r < 8; ++r) {
      const int row = mblk + wave * 16 + r + hi * 8;
      const int col = nblk + t * 16 + cn;
      const size_t idx = (size_t)row * N + col;
      float v = acc[t][r];
      if (Res) v += Res[idx];
      C[idx] = v;
    }
  }
}

// ---------------------------------------------------------------------------
// Fused causal flash attention, one wave per (batch, head, 16-query tile).
// Q,K,V bf16 [B,S,D] (head h at column offset h*DK), O f32 [B,S,D].
// ---------------------------------------------------------------------------
__global__ __launch_bounds__(32) void attention_kernel(
    const bf16_t* __restrict__ Q, const bf16_t* __restrict__ Km,
    const bf16_t* __restrict__ Vm, float* __restrict__ O)
{
  __shared__ float  pT[16 * 32];   // P tile staging (C-layout -> A-layout)
  __shared__ bf16_t vs[32 * 64];   // V tile staging (32 keys x 64 cols)

  const int bq = blockIdx.x, h = blockIdx.y, b = blockIdx.z;
  const int lane = threadIdx.x & 31;
  const int cn = lane & 15, hi = lane >> 4;

  const bf16_t* qbase = Q  + (size_t)b * S_ * D_ + h * DK;
  const bf16_t* kbase = Km + (size_t)b * S_ * D_ + h * DK;
  const bf16_t* vbase = Vm + (size_t)b * S_ * D_ + h * DK;
  const int q0 = bq * 16;

  const v16bf qa0 = load_a_frag(qbase, D_, q0, 0);
  const v16bf qa1 = load_a_frag(qbase, D_, q0, 32);

  v8f oacc[4] = {v8f{}, v8f{}, v8f{}, v8f{}};
  float mrow[8], lrow[8];
  #pragma unroll
  for (int r = 0; r < 8; ++r) { mrow[r] = -1e30f; lrow[r] = 0.f; }

  const int jmax = (q0 + 15) / 32;  // inclusive, 32 keys per step
  for (int j = 0; j <= jmax; ++j) {
    const int k0 = j * 32;

    // async-stage V tile early: lane ln owns key row (k0+ln), 64 bf16 = 8x16B
    {
      const bf16_t* vp = vbase + (size_t)(k0 + lane) * D_;
      bf16_t* dst = &vs[lane * 64];
      #pragma unroll
      for (int t = 0; t < 8; ++t) async_cp16(dst + t * 8, vp + t * 8);
    }

    // scores for 32 keys: two 16x16 C tiles
    v16bf kb0 = load_b_frag(kbase, D_, k0,      0);
    v16bf kb1 = load_b_frag(kbase, D_, k0,      32);
    v16bf kb2 = load_b_frag(kbase, D_, k0 + 16, 0);
    v16bf kb3 = load_b_frag(kbase, D_, k0 + 16, 32);
    v8f s0 = {}, s1 = {};
    s0 = wmma_bf16(qa0, kb0, s0); s0 = wmma_bf16(qa1, kb1, s0);
    s1 = wmma_bf16(qa0, kb2, s1); s1 = wmma_bf16(qa1, kb3, s1);

    // causal mask + scale, per-row online softmax update
    float sc0[8], sc1[8], rmax[8], rsum[8], esc[8];
    #pragma unroll
    for (int r = 0; r < 8; ++r) {
      const int qrow = q0 + r + hi * 8;
      sc0[r] = (k0 + cn      <= qrow) ? s0[r] * 0.125f : -1e30f;
      sc1[r] = (k0 + 16 + cn <= qrow) ? s1[r] * 0.125f : -1e30f;
      rmax[r] = fmaxf(sc0[r], sc1[r]);
    }
    #pragma unroll
    for (int off = 1; off < 16; off <<= 1) {
      #pragma unroll
      for (int r = 0; r < 8; ++r)
        rmax[r] = fmaxf(rmax[r], __shfl_xor(rmax[r], off, 32));
    }
    #pragma unroll
    for (int r = 0; r < 8; ++r) {
      const float mnew = fmaxf(mrow[r], rmax[r]);
      esc[r] = __expf(mrow[r] - mnew);
      mrow[r] = mnew;
      sc0[r] = __expf(sc0[r] - mnew);
      sc1[r] = __expf(sc1[r] - mnew);
      rsum[r] = sc0[r] + sc1[r];
    }
    #pragma unroll
    for (int off = 1; off < 16; off <<= 1) {
      #pragma unroll
      for (int r = 0; r < 8; ++r)
        rsum[r] += __shfl_xor(rsum[r], off, 32);
    }
    #pragma unroll
    for (int r = 0; r < 8; ++r) {
      lrow[r] = lrow[r] * esc[r] + rsum[r];
      #pragma unroll
      for (int t = 0; t < 4; ++t) oacc[t][r] *= esc[r];
      const int prow = r + hi * 8;
      pT[prow * 32 + cn]      = sc0[r];
      pT[prow * 32 + 16 + cn] = sc1[r];
    }
    wait_async0();     // V tile landed in LDS
    __syncthreads();

    // P: C-layout -> A-layout (16x32, K = key index) with bf16 convert
    union { v16bf v; bf16_t e[16]; } pu;
    #pragma unroll
    for (int jj = 0; jj < 8; ++jj) {
      pu.e[jj]     = (bf16_t)pT[cn * 32 + hi * 8 + jj];        // K = hi*8 + jj
      pu.e[8 + jj] = (bf16_t)pT[cn * 32 + 16 + hi * 8 + jj];   // K = 16+hi*8+jj
    }

    // V as B operand (32x16 per n-tile) from LDS, accumulate P@V
    #pragma unroll
    for (int t = 0; t < 4; ++t) {
      union { v16bf v; bf16_t e[16]; } vu;
      #pragma unroll
      for (int jj = 0; jj < 16; ++jj)
        vu.e[jj] = vs[((jj < 8 ? 0 : 8) + hi * 8 + jj) * 64 + t * 16 + cn];
      oacc[t] = wmma_bf16(pu.v, vu.v, oacc[t]);
    }
    __syncthreads();
  }

  // finalize: divide by row sums, write out
  #pragma unroll
  for (int t = 0; t < 4; ++t) {
    #pragma unroll
    for (int r = 0; r < 8; ++r) {
      const int row = q0 + r + hi * 8;
      const int col = h * DK + t * 16 + cn;
      O[((size_t)b * S_ + row) * D_ + col] = oacc[t][r] / lrow[r];
    }
  }
}

// ---------------------------------------------------------------------------
// Elementwise helpers
// ---------------------------------------------------------------------------
__global__ void embed_kernel(const int* __restrict__ ids,
                             const float* __restrict__ emb,
                             float* __restrict__ X) {
  const int row = blockIdx.x;
  const int tok = ids[row];
  const float* src = emb + (size_t)tok * D_;
  float* dst = X + (size_t)row * D_;
  for (int i = threadIdx.x; i < D_; i += blockDim.x) dst[i] = src[i];
}

__global__ __launch_bounds__(256) void rmsnorm_kernel(
    const float* __restrict__ X, const float* __restrict__ Wt,
    bf16_t* __restrict__ Out) {
  __shared__ float red[256];
  const int row = blockIdx.x;
  const float* x = X + (size_t)row * D_;
  float s = 0.f;
  for (int i = threadIdx.x; i < D_; i += 256) { const float v = x[i]; s += v * v; }
  red[threadIdx.x] = s;
  __syncthreads();
  for (int off = 128; off > 0; off >>= 1) {
    if (threadIdx.x < off) red[threadIdx.x] += red[threadIdx.x + off];
    __syncthreads();
  }
  const float inv = 1.0f / (sqrtf(red[0] / (float)D_) + 1e-5f);
  for (int i = threadIdx.x; i < D_; i += 256)
    Out[(size_t)row * D_ + i] = (bf16_t)(x[i] * inv * Wt[i]);
}

__global__ void cvt_bf16_kernel(const float* __restrict__ in,
                                bf16_t* __restrict__ out, int n) {
  const int i = blockIdx.x * blockDim.x + threadIdx.x;
  if (i < n) out[i] = (bf16_t)in[i];
}

__global__ void rope_kernel(float* __restrict__ Qf, float* __restrict__ Kf) {
  const int idx = blockIdx.x * blockDim.x + threadIdx.x;
  const int total = B_ * S_ * H_ * (DK / 2);
  if (idx >= total) return;
  const int i = idx & 31;                 // pair index in head dim
  const int hh = (idx >> 5) & (H_ - 1);
  const int s = (idx >> 9) & (S_ - 1);
  const int b = idx >> 19;
  const float inv_freq = __powf(10000.0f, -(float)i / 32.0f);
  const float ang = (float)s * inv_freq;
  float sn, c;
  __sincosf(ang, &sn, &c);
  const size_t base = ((size_t)(b * S_ + s)) * D_ + hh * DK + 2 * i;
  const float q1 = Qf[base], q2 = Qf[base + 1];
  Qf[base]     = q1 * c - q2 * sn;
  Qf[base + 1] = q1 * sn + q2 * c;
  const float k1 = Kf[base], k2 = Kf[base + 1];
  Kf[base]     = k1 * c - k2 * sn;
  Kf[base + 1] = k1 * sn + k2 * c;
}

__global__ void swiglu_kernel(const float* __restrict__ G1,
                              const float* __restrict__ G3,
                              bf16_t* __restrict__ Out, int n) {
  const int i = blockIdx.x * blockDim.x + threadIdx.x;
  if (i < n) {
    const float g = G1[i];
    const float sw = g / (1.0f + __expf(-g));
    Out[i] = (bf16_t)(sw * G3[i]);
  }
}

// ---------------------------------------------------------------------------
// Host orchestration
// ---------------------------------------------------------------------------
static inline void launch_cvt(const float* src, bf16_t* dst, size_t n,
                              hipStream_t s) {
  cvt_bf16_kernel<<<(unsigned)((n + 255) / 256), 256, 0, s>>>(src, dst, (int)n);
}
static inline void launch_gemm(const bf16_t* A, const bf16_t* W, float* C,
                               const float* Res, int M, int N, int K,
                               hipStream_t s) {
  dim3 grid((M + BM - 1) / BM, N / BN);
  gemm_bf16_kernel<<<grid, 256, 0, s>>>(A, W, C, Res, M, N, K);
}

extern "C" void kernel_launch(void* const* d_in, const int* in_sizes, int n_in,
                              void* d_out, int out_size, void* d_ws, size_t ws_size,
                              hipStream_t stream) {
  const int*   token_ids  = (const int*)d_in[0];
  const float* tok_emb    = (const float*)d_in[1];
  const float* lm_head_w  = (const float*)d_in[2];
  const float* ln_final_w = (const float*)d_in[3];
  const float* q_w        = (const float*)d_in[4];
  const float* k_w        = (const float*)d_in[5];
  const float* v_w        = (const float*)d_in[6];
  const float* o_w        = (const float*)d_in[7];
  const float* ln1_w      = (const float*)d_in[8];
  const float* ln2_w      = (const float*)d_in[9];
  const float* w1         = (const float*)d_in[10];
  const float* w2         = (const float*)d_in[11];
  const float* w3         = (const float*)d_in[12];
  float* out = (float*)d_out;
  (void)in_sizes; (void)n_in; (void)out_size; (void)ws_size;

  char* p = (char*)d_ws;
  auto take = [&](size_t bytes) -> char* {
    char* r = p;
    p += (bytes + 255) & ~(size_t)255;
    return r;
  };

  bf16_t* wbuf = (bf16_t*)take((size_t)V_ * D_ * sizeof(bf16_t)); // reused weights
  bf16_t* hbf  = (bf16_t*)take((size_t)MTOT * D_ * sizeof(bf16_t));
  float*  x    = (float*) take((size_t)MTOT * D_ * sizeof(float));
  float*  qf   = (float*) take((size_t)MTOT * D_ * sizeof(float));
  float*  kf   = (float*) take((size_t)MTOT * D_ * sizeof(float));
  float*  vf   = (float*) take((size_t)MTOT * D_ * sizeof(float));
  bf16_t* qb   = (bf16_t*)take((size_t)MTOT * D_ * sizeof(bf16_t));
  bf16_t* kb   = (bf16_t*)take((size_t)MTOT * D_ * sizeof(bf16_t));
  bf16_t* vb   = (bf16_t*)take((size_t)MTOT * D_ * sizeof(bf16_t));
  float*  ao   = (float*) take((size_t)MTOT * D_ * sizeof(float));
  bf16_t* abf  = (bf16_t*)take((size_t)MTOT * D_ * sizeof(bf16_t));
  float*  g1   = (float*) take((size_t)MTOT * F_ * sizeof(float));
  float*  g3   = (float*) take((size_t)MTOT * F_ * sizeof(float));
  bf16_t* fbf  = (bf16_t*)take((size_t)MTOT * F_ * sizeof(bf16_t));

  // token embedding gather
  embed_kernel<<<MTOT, 256, 0, stream>>>(token_ids, tok_emb, x);

  const int rope_total = B_ * S_ * H_ * (DK / 2);

  for (int l = 0; l < L_; ++l) {
    // --- attention block ---
    rmsnorm_kernel<<<MTOT, 256, 0, stream>>>(x, ln1_w + (size_t)l * D_, hbf);

    launch_cvt(q_w + (size_t)l * D_ * D_, wbuf, (size_t)D_ * D_, stream);
    launch_gemm(hbf, wbuf, qf, nullptr, MTOT, D_, D_, stream);
    launch_cvt(k_w + (size_t)l * D_ * D_, wbuf, (size_t)D_ * D_, stream);
    launch_gemm(hbf, wbuf, kf, nullptr, MTOT, D_, D_, stream);
    launch_cvt(v_w + (size_t)l * D_ * D_, wbuf, (size_t)D_ * D_, stream);
    launch_gemm(hbf, wbuf, vf, nullptr, MTOT, D_, D_, stream);

    rope_kernel<<<(rope_total + 255) / 256, 256, 0, stream>>>(qf, kf);

    launch_cvt(qf, qb, (size_t)MTOT * D_, stream);
    launch_cvt(kf, kb, (size_t)MTOT * D_, stream);
    launch_cvt(vf, vb, (size_t)MTOT * D_, stream);

    attention_kernel<<<dim3(S_ / 16, H_, B_), 32, 0, stream>>>(qb, kb, vb, ao);

    launch_cvt(ao, abf, (size_t)MTOT * D_, stream);
    launch_cvt(o_w + (size_t)l * D_ * D_, wbuf, (size_t)D_ * D_, stream);
    launch_gemm(abf, wbuf, x, x, MTOT, D_, D_, stream);  // residual add

    // --- FFN block ---
    rmsnorm_kernel<<<MTOT, 256, 0, stream>>>(x, ln2_w + (size_t)l * D_, hbf);

    launch_cvt(w1 + (size_t)l * F_ * D_, wbuf, (size_t)F_ * D_, stream);
    launch_gemm(hbf, wbuf, g1, nullptr, MTOT, F_, D_, stream);
    launch_cvt(w3 + (size_t)l * F_ * D_, wbuf, (size_t)F_ * D_, stream);
    launch_gemm(hbf, wbuf, g3, nullptr, MTOT, F_, D_, stream);

    swiglu_kernel<<<(MTOT * F_ + 255) / 256, 256, 0, stream>>>(
        g1, g3, fbf, MTOT * F_);

    launch_cvt(w2 + (size_t)l * D_ * F_, wbuf, (size_t)D_ * F_, stream);
    launch_gemm(fbf, wbuf, x, x, MTOT, D_, F_, stream);  // residual add
  }

  // final norm + lm_head
  rmsnorm_kernel<<<MTOT, 256, 0, stream>>>(x, ln_final_w, hbf);
  launch_cvt(lm_head_w, wbuf, (size_t)V_ * D_, stream);
  launch_gemm(hbf, wbuf, out, nullptr, MTOT, V_, D_, stream);
}
